// LogisticRegression_50148038148446
// MI455X (gfx1250) — matrix operand order repaired and checked by reading, not compiled, not run on hardware
//
#include <hip/hip_runtime.h>
#include <math.h>

// ---------------- problem constants (match reference) ----------------
constexpr int kN  = 64;          // samples
constexpr int kH  = 128;         // heads
constexpr int kE  = 32;          // embedding dim
constexpr int kS  = 4096;        // skill vocab
constexpr int kP  = kH * (kH - 1) / 2;   // 8128 pairs
constexpr int kNH = kN * kH;             // 8192

// feats layout: [users(32) items(32) langs(16) | onehot(4096) | tcf(8192)]
constexpr int kOnehotBase = 32 + 32 + 16;          // 80
constexpr int kTcfBase    = kOnehotBase + kS;      // 4176

typedef __attribute__((ext_vector_type(16))) _Float16 v16h;
typedef __attribute__((ext_vector_type(8)))  _Float16 h8;
typedef __attribute__((ext_vector_type(8)))  float    v8f;
typedef __attribute__((ext_vector_type(4)))  int      v4i;

union V16 { v16h v; h8 h[2]; };

#if __has_builtin(__builtin_amdgcn_global_load_async_to_lds_b128) && \
    __has_builtin(__builtin_amdgcn_s_wait_asynccnt)
#define HAS_ASYNC_LDS 1
#else
#define HAS_ASYNC_LDS 0
#endif

// p -> (i,j) of tril_indices(H,-1): p = i*(i-1)/2 + j, j<i
__device__ inline void pair_ij(int p, int& i, int& j) {
  int ii = (int)((1.0f + sqrtf(1.0f + 8.0f * (float)p)) * 0.5f);
  while (ii * (ii - 1) / 2 > p) --ii;
  while ((ii + 1) * ii / 2 <= p) ++ii;
  i = ii;
  j = p - ii * (ii - 1) / 2;
}

// ---------------- kernel 1: gather emb -> f16, transpose weights, zero accum ----------------
__global__ __launch_bounds__(256) void prep_kernel(
    const int* __restrict__ skills, const float* __restrict__ emb_table,
    const float* __restrict__ W1, const float* __restrict__ W2,
    _Float16* __restrict__ embF16, _Float16* __restrict__ W1t, _Float16* __restrict__ W2t,
    float* __restrict__ seg_total, float* __restrict__ seg_correct,
    float* __restrict__ tdot, float* __restrict__ lossacc)
{
  int idx = blockIdx.x * 256 + threadIdx.x;
  if (idx >= kNH) return;

  // emb gather + f32->f16 (row of 32, 64B)
  int sk = skills[idx];
  const float* src = emb_table + (size_t)sk * kE;
  _Float16* dst = embF16 + (size_t)idx * kE;
#pragma unroll
  for (int e = 0; e < kE; ++e) dst[e] = (_Float16)src[e];

  // W1 (64x128 row-major) -> col-major f16: W1t[c*64+k]
  { int c = idx & 127, k = idx >> 7; W1t[c * 64 + k] = (_Float16)W1[k * 128 + c]; }
  // W2 (128x64 row-major) -> col-major f16: W2t[c*128+k]
  { int c = idx & 63,  k = idx >> 6; W2t[c * 128 + k] = (_Float16)W2[k * 64 + c]; }

  seg_total[idx] = 0.0f;
  seg_correct[idx] = 0.0f;
  if (idx < kN) tdot[idx] = 0.0f;
  if (idx < 2)  lossacc[idx] = 0.0f;
}

// ---------------- kernel 2: pairwise MLP via WMMA + seg scatter ----------------
__global__ __launch_bounds__(256) void mlp_pairs_kernel(
    const _Float16* __restrict__ embF16,
    const _Float16* __restrict__ W1t, const _Float16* __restrict__ W2t,
    const float* __restrict__ b1, const float* __restrict__ b2,
    const float* __restrict__ Wsv, const float* __restrict__ bs,
    const float* __restrict__ targets,
    float* __restrict__ seg_total, float* __restrict__ seg_correct)
{
  __shared__ __align__(16) _Float16 sW1[64 * 128];        // col-major: [c*64 + k]
  __shared__ __align__(16) _Float16 sW2[128 * 64];        // col-major: [c*128 + k]
  __shared__ __align__(16) _Float16 sH1[8 * 16 * 128];    // per-wave 16x128 row-major

  const int tid   = threadIdx.x;
  const int lane  = tid & 31;
  const int wave  = tid >> 5;
  const int n     = blockIdx.x >> 6;      // 64 pair-tiles per sample
  const int tile  = blockIdx.x & 63;
  const int pbase = tile * 128 + wave * 16;

  // stage f16 weights into LDS
#if HAS_ASYNC_LDS
  // CDNA5 async DMA: global -> LDS without touching VGPRs (ASYNCcnt tracked)
#pragma unroll
  for (int k = tid; k < 1024; k += 256) {
    const v4i* g1 = (const v4i*)(W1t + 8 * k);
    const v4i* g2 = (const v4i*)(W2t + 8 * k);
    v4i* s1 = (v4i*)(sW1 + 8 * k);
    v4i* s2 = (v4i*)(sW2 + 8 * k);
    __builtin_amdgcn_global_load_async_to_lds_b128(
        (__attribute__((address_space(1))) v4i*)g1,
        (__attribute__((address_space(3))) v4i*)s1, 0, 0);
    __builtin_amdgcn_global_load_async_to_lds_b128(
        (__attribute__((address_space(1))) v4i*)g2,
        (__attribute__((address_space(3))) v4i*)s2, 0, 0);
  }
  __builtin_amdgcn_s_wait_asynccnt(0);
#else
  {
    const h8* g1 = (const h8*)W1t; h8* s1 = (h8*)sW1;
    const h8* g2 = (const h8*)W2t; h8* s2 = (h8*)sW2;
#pragma unroll
    for (int k = tid; k < 1024; k += 256) { s1[k] = g1[k]; s2[k] = g2[k]; }
  }
#endif
  __syncthreads();

  const int mrow   = lane & 15;   // row of A / column of B,C within 16-tile
  const int hiHalf = lane >> 4;   // lane group selects K sub-chunks

  // hoist biases / Ws so they clause into one load burst, independent of WMMA issue
  float bias1v[8];
#pragma unroll
  for (int nt = 0; nt < 8; ++nt) bias1v[nt] = b1[nt * 16 + mrow];
  float bias2v[4], wscv[4];
#pragma unroll
  for (int t = 0; t < 4; ++t) { bias2v[t] = b2[t * 16 + mrow]; wscv[t] = Wsv[t * 16 + mrow]; }
  const float bsv = bs[0];

  // ---- layer-1 A fragments: x = [emb_i | emb_j], 16x64 f16, 2 K-tiles of 16x32
  int p = pbase + mrow; if (p >= kP) p = kP - 1;   // clamp loads; mask at scatter
  int pi, pj; pair_ij(p, pi, pj);
  const _Float16* rowI = embF16 + (size_t)(n * kH + pi) * kE;
  const _Float16* rowJ = embF16 + (size_t)(n * kH + pj) * kE;
  V16 a0, a1;
  a0.h[0] = *(const h8*)(rowI + hiHalf * 8);        // K 0-7   / 8-15
  a0.h[1] = *(const h8*)(rowI + 16 + hiHalf * 8);   // K 16-23 / 24-31
  a1.h[0] = *(const h8*)(rowJ + hiHalf * 8);
  a1.h[1] = *(const h8*)(rowJ + 16 + hiHalf * 8);

  _Float16* myH1 = sH1 + wave * (16 * 128);

  // ---- layer 1: h1 = relu(x @ W1 + b1), 8 N-tiles x 2 K-steps of wmma
  //     C starts as inline 0 (no splat movs); bias folded into the ReLU epilogue
#pragma unroll
  for (int nt = 0; nt < 8; ++nt) {
    const int col = nt * 16 + mrow;
    v8f acc = {};                                   // -> v_wmma ..., 0 (inline C)
    const _Float16* cb = sW1 + col * 64;
    V16 bf0, bf1;
    bf0.h[0] = *(const h8*)(cb + hiHalf * 8);             // K 0-7,16-23 / 8-15,24-31
    bf0.h[1] = *(const h8*)(cb + 16 + hiHalf * 8);
    bf1.h[0] = *(const h8*)(cb + 32 + hiHalf * 8);        // K 32-39,48-55 / 40-47,56-63
    bf1.h[1] = *(const h8*)(cb + 32 + 16 + hiHalf * 8);
    acc = __builtin_amdgcn_wmma_f32_16x16x32_f16(false, a0.v, false, bf0.v,
                                                 (short)0, acc, false, false);
    acc = __builtin_amdgcn_wmma_f32_16x16x32_f16(false, a1.v, false, bf1.v,
                                                 (short)0, acc, false, false);
    // bias + relu + f16 store to per-wave LDS (C layout: lane=col, VGPR v = row v / 8+v)
    const float bias = bias1v[nt];
#pragma unroll
    for (int v = 0; v < 8; ++v) {
      float r = fmaxf(acc[v] + bias, 0.0f);
      myH1[(v + hiHalf * 8) * 128 + col] = (_Float16)r;
    }
  }
  // same-wave DS is in-order: reads below see our own writes (no barrier needed)

  // ---- layer-2 A fragments from LDS h1 (row-contiguous 16B chunks)
  V16 af[4];
#pragma unroll
  for (int kt = 0; kt < 4; ++kt) {
    const _Float16* ab = myH1 + mrow * 128 + kt * 32 + hiHalf * 8;
    af[kt].h[0] = *(const h8*)(ab);
    af[kt].h[1] = *(const h8*)(ab + 16);
  }

  // ---- layer 2: h2 = relu(h1 @ W2 + b2); fold Ws dot per tile
  float part[8] = {0.f, 0.f, 0.f, 0.f, 0.f, 0.f, 0.f, 0.f};
#pragma unroll
  for (int nt2 = 0; nt2 < 4; ++nt2) {
    const int col2 = nt2 * 16 + mrow;
    v8f acc2 = {};                                  // inline-0 C again
#pragma unroll
    for (int kt = 0; kt < 4; ++kt) {
      const _Float16* bb = sW2 + col2 * 128 + kt * 32 + hiHalf * 8;
      V16 bf;
      bf.h[0] = *(const h8*)(bb);
      bf.h[1] = *(const h8*)(bb + 16);
      acc2 = __builtin_amdgcn_wmma_f32_16x16x32_f16(false, af[kt].v, false, bf.v,
                                                    (short)0, acc2, false, false);
    }
    const float bias2 = bias2v[nt2];
    const float wsc   = wscv[nt2];
#pragma unroll
    for (int v = 0; v < 8; ++v) part[v] += fmaxf(acc2[v] + bias2, 0.0f) * wsc;
  }

  // reduce h2·Ws across the 16 lanes of each half (xor tree stays within half)
#pragma unroll
  for (int v = 0; v < 8; ++v) {
#pragma unroll
    for (int off = 1; off < 16; off <<= 1)
      part[v] += __shfl_xor(part[v], off, 32);
  }

  // scatter: lane 0 handles rows 0-7, lane 16 rows 8-15
  if (mrow == 0) {
#pragma unroll
    for (int v = 0; v < 8; ++v) {
      int row = v + hiHalf * 8;
      int pp = pbase + row;
      if (pp < kP) {
        int si, sj; pair_ij(pp, si, sj);
        float s = tanhf(part[v] + bsv);
        atomicAdd(&seg_total[n * kH + si], s);
        atomicAdd(&seg_correct[n * kH + si], s * targets[n * kH + sj]);
      }
    }
  }
}

// ---------------- kernel 3: per-sample tcf · Wlin (linearity => exact) ----------------
__global__ __launch_bounds__(32) void tdot_kernel(
    const int* __restrict__ skills, const float* __restrict__ Wlin,
    const float* __restrict__ seg_total, const float* __restrict__ seg_correct,
    float* __restrict__ tdot)
{
  const int n = blockIdx.x;
  const int lane = threadIdx.x;
  float acc = 0.0f;
#pragma unroll
  for (int h = lane; h < kH; h += 32) {
    int s = skills[n * kH + h];
    acc += seg_total[n * kH + h]   * Wlin[kTcfBase + s];
    acc += seg_correct[n * kH + h] * Wlin[kTcfBase + kS + s];
  }
#pragma unroll
  for (int off = 16; off; off >>= 1) acc += __shfl_xor(acc, off, 32);
  if (lane == 0) tdot[n] = acc;
}

// ---------------- kernel 4: logits, sigmoid, masked BCE accumulate ----------------
__global__ __launch_bounds__(256) void logits_kernel(
    const int* __restrict__ skills, const float* __restrict__ targets,
    const float* __restrict__ users, const float* __restrict__ items,
    const float* __restrict__ langs, const unsigned char* __restrict__ mask,
    const float* __restrict__ Wlin, const float* __restrict__ blin,
    const float* __restrict__ tdot,
    float* __restrict__ out, float* __restrict__ lossacc)
{
  int idx = blockIdx.x * 256 + threadIdx.x;
  if (idx >= kNH) return;
  int n = idx >> 7;

  float z = blin[0] + tdot[n];
  const float* urow = users + n * 32;
#pragma unroll
  for (int u = 0; u < 32; ++u) z += urow[u] * Wlin[u];
  const float* irow = items + (size_t)idx * 32;
#pragma unroll
  for (int k = 0; k < 32; ++k) z += irow[k] * Wlin[32 + k];
  const float* lrow = langs + (size_t)idx * 16;
#pragma unroll
  for (int l = 0; l < 16; ++l) z += lrow[l] * Wlin[64 + l];
  z += Wlin[kOnehotBase + skills[idx]];

  out[1 + idx] = 1.0f / (1.0f + expf(-z));

  float y = targets[idx];
  float m = mask[idx] ? 1.0f : 0.0f;
  float bce = fmaxf(z, 0.0f) - z * y + log1pf(expf(-fabsf(z)));
  float a = bce * m, b = m;
#pragma unroll
  for (int off = 16; off; off >>= 1) {
    a += __shfl_xor(a, off, 32);
    b += __shfl_xor(b, off, 32);
  }
  if ((threadIdx.x & 31) == 0) {
    atomicAdd(&lossacc[0], a);
    atomicAdd(&lossacc[1], b);
  }
}

__global__ void finalize_kernel(const float* __restrict__ lossacc, float* __restrict__ out) {
  out[0] = lossacc[0] / fmaxf(lossacc[1], 1.0f);
}

// ---------------- host launch ----------------
extern "C" void kernel_launch(void* const* d_in, const int* in_sizes, int n_in,
                              void* d_out, int out_size, void* d_ws, size_t ws_size,
                              hipStream_t stream) {
  (void)in_sizes; (void)n_in; (void)out_size; (void)ws_size;

  const int*   skills   = (const int*)d_in[0];
  const float* targets  = (const float*)d_in[1];
  const float* users    = (const float*)d_in[2];
  const float* items    = (const float*)d_in[3];
  const float* langs    = (const float*)d_in[4];
  const unsigned char* mask = (const unsigned char*)d_in[5];
  const float* emb_table = (const float*)d_in[6];
  const float* W1   = (const float*)d_in[7];
  const float* b1   = (const float*)d_in[8];
  const float* W2   = (const float*)d_in[9];
  const float* b2   = (const float*)d_in[10];
  const float* Ws   = (const float*)d_in[11];
  const float* bs   = (const float*)d_in[12];
  const float* Wlin = (const float*)d_in[13];
  const float* blin = (const float*)d_in[14];

  float* out = (float*)d_out;

  // workspace layout (256B aligned chunks)
  char* ws = (char*)d_ws;
  _Float16* embF16  = (_Float16*)(ws);                      // 8192*32*2   = 524288 B
  _Float16* W1t     = (_Float16*)(ws + 524288);             // 8192*2      =  16384 B
  _Float16* W2t     = (_Float16*)(ws + 540672);             // 8192*2      =  16384 B
  float* seg_total  = (float*)(ws + 557056);                // 8192*4      =  32768 B
  float* seg_correct= (float*)(ws + 589824);                // 8192*4      =  32768 B
  float* tdot       = (float*)(ws + 622592);                // 64*4
  float* lossacc    = (float*)(ws + 622848);                // 2*4

  // 1) gather/convert/zero
  prep_kernel<<<kNH / 256, 256, 0, stream>>>(
      skills, emb_table, W1, W2, embF16, W1t, W2t,
      seg_total, seg_correct, tdot, lossacc);

  // 2) WMMA pairwise MLP: 64 samples x 64 tiles of 128 pairs (8 waves x 16)
  mlp_pairs_kernel<<<kN * 64, 256, 0, stream>>>(
      embF16, W1t, W2t, b1, b2, Ws, bs, targets, seg_total, seg_correct);

  // 3) per-sample tcf dot
  tdot_kernel<<<kN, 32, 0, stream>>>(skills, Wlin, seg_total, seg_correct, tdot);

  // 4) logits + sigmoid + masked BCE accumulation
  logits_kernel<<<kNH / 256, 256, 0, stream>>>(
      skills, targets, users, items, langs, mask, Wlin, blin, tdot, out, lossacc);

  // 5) finalize loss
  finalize_kernel<<<1, 1, 0, stream>>>(lossacc, out);
}
